// NerModel_15075335209192
// MI455X (gfx1250) — matrix operand order repaired and checked by reading, not compiled, not run on hardware
//
#include <hip/hip_runtime.h>
#include <hip/hip_bf16.h>
#include <math.h>

typedef __attribute__((ext_vector_type(16))) __bf16 v16bf;
typedef __attribute__((ext_vector_type(8)))  float  v8f;

#define SEQL   256
#define HIDD   768
#define NHEAD  12
#define HDIM   64
#define FFD    3072
#define BATCH  16
#define LSTMH  256
#define NLAB   9
#define NTOK   (BATCH*SEQL)   // 4096

// ---------------------------------------------------------------- helpers
__device__ inline v8f vzero8(){ v8f z;
#pragma unroll
  for (int i=0;i<8;++i) z[i]=0.f; return z; }

__device__ inline v8f wmma_bf16(v16bf a, v16bf b, v8f c){
  return __builtin_amdgcn_wmma_f32_16x16x32_bf16(false, a, false, b, (short)0, c, false, false);
}

__device__ inline float sigf(float x){ return 1.f/(1.f+__expf(-x)); }
__device__ inline float geluf(float x){ return 0.5f*x*(1.f+erff(x*0.70710678118654752f)); }

// block of 256 threads (8 waves): sum across block, broadcast to all
__device__ inline float block_sum_256(float v, float* sbuf){
#pragma unroll
  for (int m=16;m>=1;m>>=1) v += __shfl_xor(v, m);
  int lane = threadIdx.x & 31, w = threadIdx.x >> 5;
  if (lane==0) sbuf[w] = v;
  __syncthreads();
  float r = sbuf[0];
#pragma unroll
  for (int i=1;i<8;++i) r += sbuf[i];
  __syncthreads();
  return r;
}

// ---------------------------------------------------------------- embeddings + LN
__global__ __launch_bounds__(256)
void embed_ln(const int* __restrict__ ids, const int* __restrict__ ttids,
              const float* __restrict__ wemb, const float* __restrict__ pemb,
              const float* __restrict__ temb, const float* __restrict__ g,
              const float* __restrict__ b, float* __restrict__ out)
{
  int tok = blockIdx.x;
  int s   = tok & (SEQL-1);
  int id  = ids[tok], tt = ttids[tok];
  __shared__ float sb[8];
  float v[3];
  size_t off = (size_t)tok*HIDD;
#pragma unroll
  for (int i=0;i<3;++i){
    int d = threadIdx.x + i*256;
    v[i] = wemb[(size_t)id*HIDD + d] + pemb[(size_t)s*HIDD + d] + temb[(size_t)tt*HIDD + d];
  }
  float m = block_sum_256(v[0]+v[1]+v[2], sb) * (1.f/HIDD);
  float q = 0.f;
#pragma unroll
  for (int i=0;i<3;++i){ float dd=v[i]-m; q+=dd*dd; }
  float var = block_sum_256(q, sb) * (1.f/HIDD);
  float inv = rsqrtf(var + 1e-12f);
#pragma unroll
  for (int i=0;i<3;++i){
    int d = threadIdx.x + i*256;
    out[off+d] = (v[i]-m)*inv*g[d] + b[d];
  }
}

// ---------------------------------------------------------------- residual + LN
__global__ __launch_bounds__(256)
void ln_res(const float* __restrict__ res, const float* __restrict__ x,
            const float* __restrict__ g, const float* __restrict__ b,
            float* __restrict__ out)
{
  int tok = blockIdx.x;
  __shared__ float sb[8];
  float v[3];
  size_t off = (size_t)tok*HIDD;
#pragma unroll
  for (int i=0;i<3;++i){
    int d = threadIdx.x + i*256;
    v[i] = res[off+d] + x[off+d];
  }
  float m = block_sum_256(v[0]+v[1]+v[2], sb) * (1.f/HIDD);
  float q = 0.f;
#pragma unroll
  for (int i=0;i<3;++i){ float dd=v[i]-m; q+=dd*dd; }
  float var = block_sum_256(q, sb) * (1.f/HIDD);
  float inv = rsqrtf(var + 1e-12f);
#pragma unroll
  for (int i=0;i<3;++i){
    int d = threadIdx.x + i*256;
    out[off+d] = (v[i]-m)*inv*g[d] + b[d];
  }
}

// ---------------------------------------------------------------- WMMA GEMM
// C[M,N] = act( A[M,K] @ B + bias ), A row-major f32, B row-major f32.
// b_transposed==0: B is [K,N];  ==1: B is [N,K] (compute A@B^T).
// BM=128, BN=128, BK=32, 256 threads = 8 waves (2x4), wave tile 64x32.
__global__ __launch_bounds__(256)
void gemm_bf16(const float* __restrict__ A, const float* __restrict__ B,
               const float* __restrict__ bias, float* __restrict__ C,
               int M, int N, int K, int b_transposed, int act)
{
  __shared__ __bf16 As[128][32];   // [m][k]
  __shared__ __bf16 Bs[128][32];   // [n][k] (transposed-for-B-frag)

  const int tid  = threadIdx.x;
  const int wave = tid >> 5;
  const int lane = tid & 31;
  const int wm   = wave >> 2;      // 0..1
  const int wn   = wave & 3;       // 0..3
  const int lh   = lane >> 4;      // half select
  const int l15  = lane & 15;
  const int bm0  = blockIdx.y * 128;
  const int bn0  = blockIdx.x * 128;

  v8f acc[4][2];
#pragma unroll
  for (int mt=0;mt<4;++mt)
#pragma unroll
    for (int nt=0;nt<2;++nt) acc[mt][nt] = vzero8();

  for (int k0 = 0; k0 < K; k0 += 32) {
    // stage A tile (128x32): 2 threads per row, 16 f32 each
    {
      int r = tid >> 1;
      int c = (tid & 1) * 16;
      const float* src = A + (size_t)(bm0 + r)*K + k0 + c;
#pragma unroll
      for (int i=0;i<16;++i) As[r][c+i] = (__bf16)src[i];
      if (k0 + 32 < K) __builtin_prefetch(src + 32, 0, 1);
    }
    // stage B tile as [n][k]
    if (!b_transposed) {
      int k = tid >> 3;             // 0..31
      int n0 = (tid & 7) * 16;      // 0..112
      const float* src = B + (size_t)(k0 + k)*N + bn0 + n0;
#pragma unroll
      for (int i=0;i<16;++i) Bs[n0+i][k] = (__bf16)src[i];
    } else {
      int n = tid >> 1;             // 0..127
      int ks = (tid & 1) * 16;
      const float* src = B + (size_t)(bn0 + n)*K + k0 + ks;
#pragma unroll
      for (int i=0;i<16;++i) Bs[n][ks+i] = (__bf16)src[i];
    }
    __syncthreads();

    // A-frags: 16x32 layout — lanes 0-15: K {0..7,16..23}; lanes 16-31: K {8..15,24..31}
    v16bf af[4];
#pragma unroll
    for (int mt=0;mt<4;++mt){
      int r  = wm*64 + mt*16 + l15;
      int kk = lh ? 8 : 0;
#pragma unroll
      for (int i=0;i<8;++i){ af[mt][i] = As[r][kk+i]; af[mt][8+i] = As[r][kk+16+i]; }
    }
    // B-frags: 32x16 layout — lanes 0-15: K {0..7,16..23}; lanes 16-31: K {8..15,24..31}
    v16bf bf[2];
#pragma unroll
    for (int nt=0;nt<2;++nt){
      int n  = wn*32 + nt*16 + l15;
      int kk = lh ? 8 : 0;
#pragma unroll
      for (int i=0;i<16;++i) bf[nt][i] = Bs[n][kk+i+(i&8)];
    }
#pragma unroll
    for (int mt=0;mt<4;++mt)
#pragma unroll
      for (int nt=0;nt<2;++nt)
        acc[mt][nt] = wmma_bf16(af[mt], bf[nt], acc[mt][nt]);

    __syncthreads();
  }

  // store: lane holds n = l15; VGPR r -> row r (+8 for high half)
#pragma unroll
  for (int mt=0;mt<4;++mt){
#pragma unroll
    for (int nt=0;nt<2;++nt){
      int gn = bn0 + wn*32 + nt*16 + l15;
      float bv = bias ? bias[gn] : 0.f;
#pragma unroll
      for (int r=0;r<8;++r){
        int gm = bm0 + wm*64 + mt*16 + r + (lh ? 8 : 0);
        float v = acc[mt][nt][r] + bv;
        if (act == 1) v = geluf(v);
        C[(size_t)gm*N + gn] = v;
      }
    }
  }
}

// ---------------------------------------------------------------- fused flash attention
// grid (NHEAD, BATCH); 512 threads = 16 waves; wave owns 16 query rows.
__global__ __launch_bounds__(512)
void attention(const float* __restrict__ Q, const float* __restrict__ Kd,
               const float* __restrict__ Vd, const int* __restrict__ amask,
               float* __restrict__ CTX)
{
  const int h = blockIdx.x, b = blockIdx.y;
  __shared__ __bf16 Ks[64][HDIM];       // [key][d]  — B-operand for Q@K^T
  __shared__ __bf16 Vs[HDIM][64];       // [d][key]  — B-operand for P@V
  __shared__ __bf16 Ps[16][16][64];     // per-wave probs (C-layout round-trip)
  __shared__ float  Ms[64];             // mask bias per key of the tile

  const int tid  = threadIdx.x;
  const int wave = tid >> 5;
  const int lane = tid & 31;
  const int lh   = lane >> 4;
  const int l15  = lane & 15;
  const int q0   = wave * 16;
  const size_t base = (size_t)b*SEQL*HIDD + (size_t)h*HDIM;

  // Q A-frags in registers, pre-scaled by 1/sqrt(64)
  v16bf qf[2];
  {
    const float* qs = Q + base + (size_t)(q0 + l15)*HIDD;
#pragma unroll
    for (int kb=0;kb<2;++kb){
      int kk = kb*32 + (lh ? 8 : 0);
#pragma unroll
      for (int i=0;i<8;++i){
        qf[kb][i]   = (__bf16)(qs[kk+i]    * 0.125f);
        qf[kb][8+i] = (__bf16)(qs[kk+16+i] * 0.125f);
      }
    }
  }

  float Mx[8], Ls[8];
  v8f O[4];
#pragma unroll
  for (int r=0;r<8;++r){ Mx[r] = -3e38f; Ls[r] = 0.f; }
#pragma unroll
  for (int dt=0;dt<4;++dt) O[dt] = vzero8();

  for (int kt=0; kt<SEQL/64; ++kt){
    __syncthreads();
    // stage K,V tile (64 keys x 64 dims): 8 elems/thread
    {
      int key = tid >> 3;
      int d0  = (tid & 7) * 8;
      const float* ks = Kd + base + (size_t)(kt*64 + key)*HIDD + d0;
      const float* vs = Vd + base + (size_t)(kt*64 + key)*HIDD + d0;
#pragma unroll
      for (int i=0;i<8;++i){
        Ks[key][d0+i] = (__bf16)ks[i];
        Vs[d0+i][key] = (__bf16)vs[i];
      }
    }
    if (tid < 64) Ms[tid] = (1.f - (float)amask[b*SEQL + kt*64 + tid]) * -1e9f;
    __syncthreads();

    // scores S = Q @ K^T (16 x 64)
    v8f Sacc[4];
#pragma unroll
    for (int nt=0;nt<4;++nt) Sacc[nt] = vzero8();
#pragma unroll
    for (int nt=0;nt<4;++nt){
      int n = nt*16 + l15;
#pragma unroll
      for (int kb=0;kb<2;++kb){
        v16bf bf;
        int kk = kb*32 + (lh ? 8 : 0);
#pragma unroll
        for (int i=0;i<16;++i) bf[i] = Ks[n][kk+i+(i&8)];
        Sacc[nt] = wmma_bf16(qf[kb], bf, Sacc[nt]);
      }
      float mb = Ms[n];
#pragma unroll
      for (int r=0;r<8;++r) Sacc[nt][r] += mb;
    }

    // online softmax (row reductions across 16-lane halves)
#pragma unroll
    for (int r=0;r<8;++r){
      float mx = -3e38f;
#pragma unroll
      for (int nt=0;nt<4;++nt){
        float t = Sacc[nt][r];
        t = fmaxf(t, __shfl_xor(t,1));
        t = fmaxf(t, __shfl_xor(t,2));
        t = fmaxf(t, __shfl_xor(t,4));
        t = fmaxf(t, __shfl_xor(t,8));
        mx = fmaxf(mx, t);
      }
      float mnew = fmaxf(Mx[r], mx);
      float corr = __expf(Mx[r] - mnew);
      Mx[r] = mnew;
      Ls[r] *= corr;
#pragma unroll
      for (int dt=0;dt<4;++dt) O[dt][r] *= corr;
      float rs = 0.f;
#pragma unroll
      for (int nt=0;nt<4;++nt){
        float p = __expf(Sacc[nt][r] - mnew);
        Sacc[nt][r] = p;
        float t = p;
        t += __shfl_xor(t,1); t += __shfl_xor(t,2);
        t += __shfl_xor(t,4); t += __shfl_xor(t,8);
        rs += t;
      }
      Ls[r] += rs;
    }

    // C-layout -> A-layout via wave-private LDS (LDS is in-order per wave)
#pragma unroll
    for (int nt=0;nt<4;++nt)
#pragma unroll
      for (int r=0;r<8;++r)
        Ps[wave][r + (lh ? 8 : 0)][nt*16 + l15] = (__bf16)Sacc[nt][r];

    v16bf pf[2];
#pragma unroll
    for (int kb=0;kb<2;++kb){
      int kk = kb*32 + (lh ? 8 : 0);
#pragma unroll
      for (int i=0;i<8;++i){
        pf[kb][i]   = Ps[wave][l15][kk+i];
        pf[kb][8+i] = Ps[wave][l15][kk+16+i];
      }
    }
    // O += P @ V
#pragma unroll
    for (int dt=0;dt<4;++dt){
      int n = dt*16 + l15;
#pragma unroll
      for (int kb=0;kb<2;++kb){
        v16bf vf;
        int kk = kb*32 + (lh ? 8 : 0);
#pragma unroll
        for (int i=0;i<16;++i) vf[i] = Vs[n][kk+i+(i&8)];
        O[dt] = wmma_bf16(pf[kb], vf, O[dt]);
      }
    }
  }

  // normalize and store context
#pragma unroll
  for (int dt=0;dt<4;++dt){
    int d = dt*16 + l15;
#pragma unroll
    for (int r=0;r<8;++r){
      int row = q0 + r + (lh ? 8 : 0);
      CTX[base + (size_t)row*HIDD + d] = O[dt][r] / Ls[r];
    }
  }
}

// ---------------------------------------------------------------- LSTM recurrence
// grid=2 (dir), 1024 threads = 32 waves. Gates = gx[t] + h @ Whh^T + bhh.
// Per-step 16x1024x256 GEMM on WMMA, h kept bf16 in LDS, c in registers.
__global__ __launch_bounds__(1024)
void lstm_rec(const float* __restrict__ gxf, const float* __restrict__ gxb,
              const float* __restrict__ whhf, const float* __restrict__ whhb,
              const float* __restrict__ bhhf, const float* __restrict__ bhhb,
              float* __restrict__ outf, float* __restrict__ outb)
{
  const int dir = blockIdx.x;
  const float* gx  = dir ? gxb  : gxf;
  const float* whh = dir ? whhb : whhf;
  const float* bhh = dir ? bhhb : bhhf;
  float* out       = dir ? outb : outf;

  __shared__ __bf16 hS[16][LSTMH];       // 8KB
  __shared__ __bf16 gS[16][4*LSTMH];     // 32KB

  const int tid  = threadIdx.x;
  const int wave = tid >> 5;
  const int lane = tid & 31;
  const int lh   = lane >> 4;
  const int l15  = lane & 15;

  for (int i = tid; i < 16*LSTMH; i += 1024) ((__bf16*)hS)[i] = (__bf16)0.f;
  float c4[4] = {0.f, 0.f, 0.f, 0.f};
  __syncthreads();

  for (int t=0; t<SEQL; ++t){
    const int te = dir ? (SEQL-1-t) : t;
    v8f acc[2]; acc[0] = vzero8(); acc[1] = vzero8();
#pragma unroll
    for (int kb=0; kb<8; ++kb){
      v16bf af;
      int kk = kb*32 + (lh ? 8 : 0);
#pragma unroll
      for (int i=0;i<8;++i){ af[i] = hS[l15][kk+i]; af[8+i] = hS[l15][kk+16+i]; }
#pragma unroll
      for (int nt=0; nt<2; ++nt){
        int n = (wave*2 + nt)*16 + l15;
        const float* wsrc = whh + (size_t)n*LSTMH + kb*32 + (lh ? 8 : 0);
        v16bf bf;
#pragma unroll
        for (int i=0;i<16;++i) bf[i] = (__bf16)wsrc[i+(i&8)];
        acc[nt] = wmma_bf16(af, bf, acc[nt]);
      }
    }
#pragma unroll
    for (int nt=0; nt<2; ++nt){
      int n = (wave*2 + nt)*16 + l15;
#pragma unroll
      for (int r=0;r<8;++r){
        int bb = r + (lh ? 8 : 0);
        float gval = acc[nt][r] + gx[((size_t)bb*SEQL + te)*(4*LSTMH) + n] + bhh[n];
        gS[bb][n] = (__bf16)gval;
      }
    }
    __syncthreads();
#pragma unroll
    for (int i=0;i<4;++i){
      int cell = tid*4 + i;
      int bb = cell >> 8, j = cell & 255;
      float gi = (float)gS[bb][j];
      float gf = (float)gS[bb][256+j];
      float gg = (float)gS[bb][512+j];
      float go = (float)gS[bb][768+j];
      float c  = sigf(gf)*c4[i] + sigf(gi)*tanhf(gg);
      float hh = sigf(go)*tanhf(c);
      c4[i] = c;
      hS[bb][j] = (__bf16)hh;
      out[((size_t)bb*SEQL + te)*LSTMH + j] = hh;
    }
    __syncthreads();
  }
}

// ---------------------------------------------------------------- emissions
__global__ void emit_kernel(const float* __restrict__ lf, const float* __restrict__ lb,
                            const float* __restrict__ lw, const float* __restrict__ lbias,
                            float* __restrict__ emit)
{
  int idx = blockIdx.x*blockDim.x + threadIdx.x;
  if (idx >= NTOK*NLAB) return;
  int tok = idx / NLAB, lab = idx % NLAB;
  const float* f  = lf + (size_t)tok*LSTMH;
  const float* bw = lb + (size_t)tok*LSTMH;
  float s = lbias[lab];
  for (int k=0;k<LSTMH;++k) s += f[k]  * lw[k*NLAB + lab];
  for (int k=0;k<LSTMH;++k) s += bw[k] * lw[(LSTMH+k)*NLAB + lab];
  emit[idx] = s;
}

// ---------------------------------------------------------------- CRF: Viterbi + log-likelihood
__global__ __launch_bounds__(32)
void crf_kernel(const float* __restrict__ emit, const int* __restrict__ labels,
                const float* __restrict__ start, const float* __restrict__ trans,
                const float* __restrict__ endw, float* __restrict__ out_tags,
                float* __restrict__ llh)
{
  const int b = blockIdx.x;
  const int j = threadIdx.x;
  __shared__ int   hist[SEQL-1][NLAB];
  __shared__ float sc[NLAB];
  __shared__ float tr[NLAB*NLAB];
  for (int i=j; i<NLAB*NLAB; i+=32) tr[i] = trans[i];
  __syncthreads();

  const float* E = emit + (size_t)b*SEQL*NLAB;

  // ---- Viterbi
  float score = 0.f;
  if (j < NLAB) score = start[j] + E[j];
  for (int t=1; t<SEQL; ++t){
    if (j < NLAB) sc[j] = score;
    __syncthreads();
    if (j < NLAB){
      float best = -3e38f; int bi = 0;
      for (int i=0;i<NLAB;++i){
        float v = sc[i] + tr[i*NLAB + j];
        if (v > best){ best = v; bi = i; }
      }
      score = best + E[t*NLAB + j];
      hist[t-1][j] = bi;
    }
    __syncthreads();
  }
  if (j < NLAB) sc[j] = score + endw[j];
  __syncthreads();
  if (j == 0){
    float best = -3e38f; int bt = 0;
    for (int i=0;i<NLAB;++i) if (sc[i] > best){ best = sc[i]; bt = i; }
    int tag = bt;
    out_tags[b*SEQL + SEQL-1] = (float)tag;
    for (int t=SEQL-2; t>=0; --t){
      tag = hist[t][tag];
      out_tags[b*SEQL + t] = (float)tag;
    }
  }
  __syncthreads();

  // ---- forward algorithm (denominator)
  float alpha = 0.f;
  if (j < NLAB) alpha = start[j] + E[j];
  for (int t=1; t<SEQL; ++t){
    if (j < NLAB) sc[j] = alpha;
    __syncthreads();
    if (j < NLAB){
      float m = -3e38f;
      for (int i=0;i<NLAB;++i) m = fmaxf(m, sc[i] + tr[i*NLAB + j]);
      float s = 0.f;
      for (int i=0;i<NLAB;++i) s += __expf(sc[i] + tr[i*NLAB + j] - m);
      alpha = m + __logf(s) + E[t*NLAB + j];
    }
    __syncthreads();
  }
  if (j < NLAB) sc[j] = alpha + endw[j];
  __syncthreads();
  if (j == 0){
    float m = -3e38f;
    for (int i=0;i<NLAB;++i) m = fmaxf(m, sc[i]);
    float s = 0.f;
    for (int i=0;i<NLAB;++i) s += __expf(sc[i] - m);
    float denom = m + __logf(s);
    const int* lb = labels + b*SEQL;
    float num = start[lb[0]] + E[lb[0]];
    for (int t=1; t<SEQL; ++t)
      num += E[t*NLAB + lb[t]] + tr[lb[t-1]*NLAB + lb[t]];
    num += endw[lb[SEQL-1]];
    llh[b] = num - denom;
  }
}

__global__ void finalize_kernel(const float* __restrict__ llh, float* __restrict__ dout){
  if (blockIdx.x == 0 && threadIdx.x == 0){
    float s = 0.f;
    for (int i=0;i<BATCH;++i) s += llh[i];
    dout[NTOK] = -s / (float)BATCH;
  }
}

// ---------------------------------------------------------------- host orchestration
extern "C" void kernel_launch(void* const* d_in, const int* in_sizes, int n_in,
                              void* d_out, int out_size, void* d_ws, size_t ws_size,
                              hipStream_t stream)
{
  (void)in_sizes; (void)n_in; (void)out_size; (void)ws_size;
  const int*   ids    = (const int*)d_in[0];
  const int*   amask  = (const int*)d_in[1];
  const int*   ttids  = (const int*)d_in[2];
  const int*   labels = (const int*)d_in[3];
  const float* wemb   = (const float*)d_in[4];
  const float* pemb   = (const float*)d_in[5];
  const float* temb   = (const float*)d_in[6];
  const float* eg     = (const float*)d_in[7];
  const float* ebi    = (const float*)d_in[8];
  const float* qw  = (const float*)d_in[9],  *qb  = (const float*)d_in[10];
  const float* kw  = (const float*)d_in[11], *kbv = (const float*)d_in[12];
  const float* vw  = (const float*)d_in[13], *vbv = (const float*)d_in[14];
  const float* ow  = (const float*)d_in[15], *obv = (const float*)d_in[16];
  const float* l1g = (const float*)d_in[17], *l1b = (const float*)d_in[18];
  const float* f1w = (const float*)d_in[19], *f1b = (const float*)d_in[20];
  const float* f2w = (const float*)d_in[21], *f2b = (const float*)d_in[22];
  const float* l2g = (const float*)d_in[23], *l2b = (const float*)d_in[24];
  const float* wihf = (const float*)d_in[25], *whhf = (const float*)d_in[26];
  const float* bihf = (const float*)d_in[27], *bhhf = (const float*)d_in[28];
  const float* wihb = (const float*)d_in[29], *whhb = (const float*)d_in[30];
  const float* bihb = (const float*)d_in[31], *bhhb = (const float*)d_in[32];
  const float* linw = (const float*)d_in[33], *linb = (const float*)d_in[34];
  const float* cstart = (const float*)d_in[35];
  const float* cend   = (const float*)d_in[36];
  const float* ctrans = (const float*)d_in[37];

  // workspace carve (floats): 5 x [4096,768] + [4096,3072]  (~113 MB)
  float* ws  = (float*)d_ws;
  const size_t NT = (size_t)NTOK;
  float* A   = ws;                 // current hidden
  float* B1  = A   + NT*HIDD;
  float* B2  = B1  + NT*HIDD;
  float* B3  = B2  + NT*HIDD;
  float* B4  = B3  + NT*HIDD;
  float* FFB = B4  + NT*HIDD;      // 4096*3072
  float* GXF = FFB;                // reuse after BERT
  float* GXB = FFB + NT*(4*LSTMH);
  float* LF  = B1;
  float* LB  = B2;
  float* EMI = B3;
  float* LLH = B4;

  auto GEMM = [&](const float* Ap, const float* Bp, const float* bias, float* Cp,
                  int M, int N, int K, int btr, int act){
    dim3 g(N/128, M/128);
    gemm_bf16<<<g, 256, 0, stream>>>(Ap, Bp, bias, Cp, M, N, K, btr, act);
  };

  embed_ln<<<NTOK, 256, 0, stream>>>(ids, ttids, wemb, pemb, temb, eg, ebi, A);

  for (int l=0; l<12; ++l){
    const size_t w2  = (size_t)l*HIDD*HIDD;
    const size_t b1  = (size_t)l*HIDD;
    GEMM(A,  qw + w2, qb  + b1, B1, NTOK, HIDD, HIDD, 0, 0);     // Q
    GEMM(A,  kw + w2, kbv + b1, B2, NTOK, HIDD, HIDD, 0, 0);     // K
    GEMM(A,  vw + w2, vbv + b1, B3, NTOK, HIDD, HIDD, 0, 0);     // V
    attention<<<dim3(NHEAD, BATCH), 512, 0, stream>>>(B1, B2, B3, amask, B4);
    GEMM(B4, ow + w2, obv + b1, B1, NTOK, HIDD, HIDD, 0, 0);     // O-proj
    ln_res<<<NTOK, 256, 0, stream>>>(A, B1, l1g + b1, l1b + b1, B2);
    GEMM(B2, f1w + (size_t)l*HIDD*FFD, f1b + (size_t)l*FFD, FFB, NTOK, FFD, HIDD, 0, 1); // GELU
    GEMM(FFB, f2w + (size_t)l*FFD*HIDD, f2b + b1, B3, NTOK, HIDD, FFD, 0, 0);
    ln_res<<<NTOK, 256, 0, stream>>>(B2, B3, l2g + b1, l2b + b1, A);
  }

  // BiLSTM: hoist x @ Wih^T (+bih) as big GEMMs, then persistent recurrence
  GEMM(A, wihf, bihf, GXF, NTOK, 4*LSTMH, HIDD, 1, 0);
  GEMM(A, wihb, bihb, GXB, NTOK, 4*LSTMH, HIDD, 1, 0);
  lstm_rec<<<2, 1024, 0, stream>>>(GXF, GXB, whhf, whhb, bhhf, bhhb, LF, LB);

  emit_kernel<<<(NTOK*NLAB + 255)/256, 256, 0, stream>>>(LF, LB, linw, linb, EMI);

  crf_kernel<<<BATCH, 32, 0, stream>>>(EMI, labels, cstart, ctrans, cend,
                                       (float*)d_out, LLH);
  finalize_kernel<<<1, 32, 0, stream>>>(LLH, (float*)d_out);
}
